// UnifiedDecoder_77352361001492
// MI455X (gfx1250) — compile-verified, matrix operand order checked by reading, not run on hardware
//
#include <hip/hip_runtime.h>
#include <hip/hip_bf16.h>

// ---------------------------------------------------------------------------
// UnifiedDecoder GRU scan for gfx1250 (MI455X): bf16 WMMA 16x16x32, f32 accum.
// B=2048, T=100, H=256, L=128, NV=5.
// 16 wave32s per workgroup, one 16-col N-tile per wave: low register pressure
// (no spills), weights streamed from L2 (per-wave 24KB/step, L0/L1-resident).
// ---------------------------------------------------------------------------

#define B_  2048
#define T_  100
#define H_  256
#define L_  128
#define NV_ 5
#define NTHREADS 512   // 16 waves

typedef __attribute__((ext_vector_type(16))) __bf16 v16bf;
typedef __attribute__((ext_vector_type(8)))  __bf16 v8bf;
typedef __attribute__((ext_vector_type(8)))  float  v8f;

static __device__ __forceinline__ v8f wmma_bf16(v16bf a, v16bf b, v8f c) {
    // D = A(16x32 bf16) x B(32x16 bf16) + C(16x16 f32)
    return __builtin_amdgcn_wmma_f32_16x16x32_bf16(
        /*neg_a=*/false, a, /*neg_b=*/false, b,
        /*c_mod=*/(short)0, c, /*reuse_a=*/false, /*reuse_b=*/false);
}

// Load one 16x32 bf16 A-operand chunk (k-chunk kc) from a row-major LDS
// buffer of row stride `stride` (elements).  Lane l%16 = row M; lanes 0-15
// cover K {0..7,16..23}, lanes 16-31 cover K {8..15,24..31}: two contiguous
// 16B segments per lane -> two ds_load_b128.
static __device__ __forceinline__ v16bf load_a_lds(const __bf16* buf, int stride,
                                                   int kc, int lane) {
    const int row  = lane & 15;
    const int half = lane >> 4;
    const __bf16* p = buf + row * stride + kc * 32 + half * 8;
    v8bf lo = *(const v8bf*)(p);
    v8bf hi = *(const v8bf*)(p + 16);
    return __builtin_shufflevector(lo, hi, 0,1,2,3,4,5,6,7,8,9,10,11,12,13,14,15);
}

// ---------------------------------------------------------------------------
// Pack a f32 weight matrix (rows [rowoff, rowoff+32*ktmax), 256 cols) into the
// WMMA B-operand lane layout as bf16.  One 32x16 block per (kt, nt) = 512
// elements; lane l's 16 elements are contiguous (32B/lane):
//   k = kt*32 + (e&7) + ((e>>3)<<4) + ((l>>4)<<3),  n = nt*16 + (l&15)
// ---------------------------------------------------------------------------
__global__ __launch_bounds__(256) void pack_b_kernel(const float* __restrict__ src,
                                                     int rowoff, __bf16* __restrict__ dst) {
    const int gid    = blockIdx.x * 256 + threadIdx.x;
    const int blk    = gid >> 9;           // (kt*16 + nt)
    const int within = gid & 511;
    const int kt = blk >> 4, nt = blk & 15;
    const int l  = within >> 4, e = within & 15;
    const int k  = kt * 32 + (e & 7) + ((e >> 3) << 4) + ((l >> 4) << 3);
    const int n  = nt * 16 + (l & 15);
    dst[gid] = (__bf16)src[(size_t)(rowoff + k) * H_ + n];
}

// ---------------------------------------------------------------------------
// Fill the constant tree/graph halves of all_hidden_states (pure bandwidth).
// ---------------------------------------------------------------------------
__global__ __launch_bounds__(256) void fill_latents_kernel(
    const float* __restrict__ tree, const float* __restrict__ graph,
    const int* __restrict__ lengths, float* __restrict__ out_all) {
    const size_t idx = (size_t)blockIdx.x * 256 + threadIdx.x;   // B*T*2L
    if (idx >= (size_t)B_ * T_ * (2 * L_)) return;
    const int    c  = (int)(idx & 255);
    const size_t bt = idx >> 8;
    const int    t  = (int)(bt % T_);
    const int    b  = (int)(bt / T_);
    float v = 0.0f;
    if (t < lengths[b])
        v = (c < L_) ? tree[(size_t)b * L_ + c] : graph[(size_t)b * L_ + (c - L_)];
    out_all[bt * (size_t)(H_ + 2 * L_) + H_ + c] = v;
}

// ---------------------------------------------------------------------------
// Main GRU scan.  One workgroup per 16 batch rows; 16 waves, wave w owns
// output columns [16w, 16w+16) (one 16x16 N-tile) of all gates.
// ---------------------------------------------------------------------------
__global__ __launch_bounds__(NTHREADS, 1) void gru_scan_kernel(
    const float* __restrict__ x_seq,   const float* __restrict__ hidden,
    const float* __restrict__ graph,   const int*   __restrict__ lengths,
    const float* __restrict__ b_cs,
    const float* __restrict__ Wz, const float* __restrict__ b_z,
    const float* __restrict__ Wr, const float* __restrict__ b_r,
    const float* __restrict__ Wh, const float* __restrict__ b_h,
    const __bf16* __restrict__ wcs_p, const __bf16* __restrict__ wz_p,
    const __bf16* __restrict__ wr_p,  const __bf16* __restrict__ wh_p,
    float* __restrict__ out_newh, float* __restrict__ out_all,
    float* __restrict__ out_seg) {

    __shared__ __bf16 catbuf[16 * 384];     // [hidden | graph] bf16, row-major
    __shared__ __bf16 hbuf[16 * 256];       // current h, bf16 row-major
    __shared__ __bf16 rhbuf[16 * 256];      // r*h, bf16 row-major
    __shared__ float  xbuf[2][16 * NV_];    // x(t) tile, double buffered
    __shared__ float  wxbuf[3 * NV_ * H_];  // x-part weight rows [g][j][n]

    const int tid   = threadIdx.x;
    const int lane  = tid & 31;
    const int wave  = tid >> 5;             // 0..15 -> one N-tile each
    const int b0    = blockIdx.x * 16;
    const int halfm = lane >> 4;            // 0/1 -> M offset 0/8 in C layout
    const int nl    = lane & 15;

    // --- stage concat([hidden, graph]) as bf16, and x-part weight rows ---
    for (int i = tid; i < 16 * 384; i += NTHREADS) {
        const int m = i / 384, c = i % 384;
        const float v = (c < H_) ? hidden[(size_t)(b0 + m) * H_ + c]
                                 : graph[(size_t)(b0 + m) * L_ + (c - H_)];
        catbuf[i] = (__bf16)v;
    }
    for (int i = tid; i < 3 * NV_ * H_; i += NTHREADS) {
        const int g = i / (NV_ * H_);
        const int r = i % (NV_ * H_);       // j*H + n
        const float* W = (g == 0) ? Wz : (g == 1) ? Wr : Wh;
        wxbuf[i] = W[r];                    // rows 0..NV-1 are the x-part
    }
    __syncthreads();

    const int nt = wave;
    const int n  = nt * 16 + nl;

    // --- h0 = relu(cat @ W_cs + b_cs), K = 384 = 12 chunks (streamed) ---
    v8f acc = {};
    {
        const float bc = b_cs[n];
        #pragma unroll
        for (int v = 0; v < 8; ++v) acc[v] = bc;
    }
    #pragma unroll 4
    for (int kc = 0; kc < 12; ++kc) {
        const v16bf a = load_a_lds(catbuf, 384, kc, lane);
        const v16bf w = *(const v16bf*)(wcs_p + (size_t)(kc * 16 + nt) * 512 + lane * 16);
        acc = wmma_bf16(a, w, acc);
    }
    v8f hc;
    #pragma unroll
    for (int v = 0; v < 8; ++v) hc[v] = fmaxf(acc[v], 0.0f);
    #pragma unroll
    for (int v = 0; v < 8; ++v)
        hbuf[(v + 8 * halfm) * 256 + n] = (__bf16)hc[v];

    // --- per-lane constants ---
    const float bz = b_z[n], br = b_r[n], bh = b_h[n];
    int len[8];
    #pragma unroll
    for (int v = 0; v < 8; ++v) len[v] = lengths[b0 + v + 8 * halfm];

    v8f mx;
    #pragma unroll
    for (int v = 0; v < 8; ++v) mx[v] = -__builtin_inff();

    // ======================= scan over time =======================
    #pragma unroll 1
    for (int t = 0; t < T_; ++t) {
        const int pb = t & 1;
        if (tid < 16 * NV_) {
            const int m = tid / NV_, j = tid % NV_;
            xbuf[pb][tid] = x_seq[((size_t)(b0 + m) * T_ + t) * NV_ + j];
        }
        __syncthreads();   // x(t) + hbuf published

        // --- gates z, r: bias + x-part, then 8 WMMA k-steps ---
        v8f az, ar;
        #pragma unroll
        for (int v = 0; v < 8; ++v) {
            const int m = v + 8 * halfm;
            float sz = bz, sr = br;
            #pragma unroll
            for (int j = 0; j < NV_; ++j) {
                const float xv = xbuf[pb][m * NV_ + j];
                sz += xv * wxbuf[0 * NV_ * H_ + j * H_ + n];
                sr += xv * wxbuf[1 * NV_ * H_ + j * H_ + n];
            }
            az[v] = sz; ar[v] = sr;
        }
        #pragma unroll 4
        for (int kc = 0; kc < 8; ++kc) {
            const v16bf a  = load_a_lds(hbuf, 256, kc, lane);
            const v16bf wz_b = *(const v16bf*)(wz_p + (size_t)(kc * 16 + nt) * 512 + lane * 16);
            const v16bf wr_b = *(const v16bf*)(wr_p + (size_t)(kc * 16 + nt) * 512 + lane * 16);
            az = wmma_bf16(a, wz_b, az);
            ar = wmma_bf16(a, wr_b, ar);
        }
        v8f z;
        #pragma unroll
        for (int v = 0; v < 8; ++v) {
            z[v] = 1.0f / (1.0f + __expf(-az[v]));
            const float r = 1.0f / (1.0f + __expf(-ar[v]));
            rhbuf[(v + 8 * halfm) * 256 + n] = (__bf16)(r * hc[v]);
        }
        __syncthreads();   // rhbuf published; hbuf reads retired

        // --- candidate: tanh([x, r*h] @ W_h + b_h) ---
        v8f ah;
        #pragma unroll
        for (int v = 0; v < 8; ++v) {
            const int m = v + 8 * halfm;
            float s = bh;
            #pragma unroll
            for (int j = 0; j < NV_; ++j)
                s += xbuf[pb][m * NV_ + j] * wxbuf[2 * NV_ * H_ + j * H_ + n];
            ah[v] = s;
        }
        #pragma unroll 4
        for (int kc = 0; kc < 8; ++kc) {
            const v16bf a    = load_a_lds(rhbuf, 256, kc, lane);
            const v16bf wh_b = *(const v16bf*)(wh_p + (size_t)(kc * 16 + nt) * 512 + lane * 16);
            ah = wmma_bf16(a, wh_b, ah);
        }

        // --- h update + epilogues (masked stream-out, max-pool, gather) ---
        #pragma unroll
        for (int v = 0; v < 8; ++v) {
            const int m = v + 8 * halfm;
            const float p  = tanhf(ah[v]);
            const float nh = (1.0f - z[v]) * hc[v] + z[v] * p;
            hc[v] = nh;
            hbuf[m * 256 + n] = (__bf16)nh;
            const bool valid = (t < len[v]);
            const size_t rowo = ((size_t)(b0 + m) * T_ + t) * (size_t)(H_ + 2 * L_);
            out_all[rowo + n] = valid ? nh : 0.0f;
            if (valid) mx[v] = fmaxf(mx[v], nh);
            if (t == len[v] - 1)
                out_newh[(size_t)(b0 + m) * H_ + n] = nh;
        }
        // next iteration's top barrier orders hbuf/rhbuf reuse; x is
        // double-buffered so no trailing barrier is needed.
    }

    #pragma unroll
    for (int v = 0; v < 8; ++v)
        out_seg[(size_t)(b0 + v + 8 * halfm) * H_ + n] = mx[v];
}

// ---------------------------------------------------------------------------
extern "C" void kernel_launch(void* const* d_in, const int* in_sizes, int n_in,
                              void* d_out, int out_size, void* d_ws, size_t ws_size,
                              hipStream_t stream) {
    const float* x_seq  = (const float*)d_in[0];
    const float* hidden = (const float*)d_in[1];
    const float* tree   = (const float*)d_in[2];
    const float* graph  = (const float*)d_in[3];
    const int*   lens   = (const int*)d_in[4];
    const float* W_cs   = (const float*)d_in[5];
    const float* b_cs   = (const float*)d_in[6];
    const float* W_z    = (const float*)d_in[7];
    const float* b_z    = (const float*)d_in[8];
    const float* W_r    = (const float*)d_in[9];
    const float* b_r    = (const float*)d_in[10];
    const float* W_h    = (const float*)d_in[11];
    const float* b_h    = (const float*)d_in[12];

    float* out      = (float*)d_out;
    float* out_newh = out;
    float* out_all  = out + (size_t)B_ * H_;
    float* out_seg  = out_all + (size_t)B_ * T_ * (H_ + 2 * L_);

    // Workspace: packed bf16 B-operand weights (~576 KB).
    __bf16* wcs_p = (__bf16*)d_ws;            // 12 kt * 16 nt * 512 = 98304
    __bf16* wz_p  = wcs_p + 98304;            // 8 kt  * 16 nt * 512 = 65536
    __bf16* wr_p  = wz_p + 65536;
    __bf16* wh_p  = wr_p + 65536;

    // Pack weights into WMMA B-operand layout (bf16).
    pack_b_kernel<<<12 * 32, 256, 0, stream>>>(W_cs, 0, wcs_p);   // K=384
    pack_b_kernel<<< 8 * 32, 256, 0, stream>>>(W_z, NV_, wz_p);   // h-part K=256
    pack_b_kernel<<< 8 * 32, 256, 0, stream>>>(W_r, NV_, wr_p);
    pack_b_kernel<<< 8 * 32, 256, 0, stream>>>(W_h, NV_, wh_p);

    // Constant latent halves of all_hidden_states (bandwidth-bound).
    const size_t nfill = (size_t)B_ * T_ * (2 * L_);
    fill_latents_kernel<<<(unsigned)((nfill + 255) / 256), 256, 0, stream>>>(
        tree, graph, lens, out_all);

    // GRU scan: one WG per 16 batch rows, 16 wave32s each (one N-tile/wave).
    gru_scan_kernel<<<B_ / 16, NTHREADS, 0, stream>>>(
        x_seq, hidden, graph, lens, b_cs,
        W_z, b_z, W_r, b_r, W_h, b_h,
        wcs_p, wz_p, wr_p, wh_p,
        out_newh, out_all, out_seg);
}
